// CharLSTM_62809601736981
// MI455X (gfx1250) — matrix operand order repaired and checked by reading, not compiled, not run on hardware
//
#include <hip/hip_runtime.h>
#include <hip/hip_bf16.h>

// ---------------------------------------------------------------------------
// CharLSTM forward for MI455X (gfx1250), bf16 WMMA everywhere.
//   B=64, T=512, H=1024, L=2, V=256
// embed-gather+cast -> weight repack (ISA B-tile layout) -> persistent
// recurrent kernel: per step, z=[x|h]@[Wi;Wh] via v_wmma_f32_16x16x32_bf16
// with the 64-row A slab staged in LDS (shared by all 8 waves of the WG),
// grid barrier between GEMM/gate phases, per-step fused vocab projection.
// ---------------------------------------------------------------------------

typedef __bf16 bf16_t;
typedef __attribute__((ext_vector_type(16))) __bf16 v16bf;
typedef __attribute__((ext_vector_type(8)))  __bf16 v8bf;
typedef __attribute__((ext_vector_type(8)))  float  v8f;

#define B_   64
#define T_   512
#define H_   1024
#define G4_  4096   // 4*H
#define V_   256
#define NWG  64
#define NTHR 256
#define NWAVE_WG 8
#define GRID_THREADS (NWG * NTHR)      // 16384

// LDS staging: one K-chunk slab = 64 rows x 32 bf16, padded 64B->80B per row
// (row starts hit banks 0,20,40,..: conflict-free b128 access pattern).
#define ROWPAD   80
#define SLAB     (B_ * ROWPAD)         // 5120 B
#define KGROUP   4                     // K-chunks staged per barrier
#define SMEM_BYTES (2 * KGROUP * SLAB) // double buffered: 40 KB

__device__ __forceinline__ bf16_t f2bf(float f) {
    unsigned u = __float_as_uint(f);
    u += 0x7FFFu + ((u >> 16) & 1u);            // round-to-nearest-even
    unsigned short s = (unsigned short)(u >> 16);
    bf16_t r;
    __builtin_memcpy(&r, &s, 2);
    return r;
}

__device__ __forceinline__ float sigf(float x) { return 1.0f / (1.0f + __expf(-x)); }

// -------------------------- grid-wide barrier ------------------------------
// Monotonic arrival counter (no reset race): barrier #target completes when
// total arrivals == target*NWG. gen is bumped by the last arriver.
__device__ __forceinline__ void grid_sync(unsigned* bar, unsigned* gen, unsigned target) {
    __threadfence();
    __syncthreads();
    if (threadIdx.x == 0) {
        unsigned arrive = atomicAdd(bar, 1u) + 1u;
        if (arrive == target * NWG) {
            atomicAdd(gen, 1u);
        } else {
            while (__hip_atomic_load(gen, __ATOMIC_ACQUIRE, __HIP_MEMORY_SCOPE_AGENT) < target) {
                __builtin_amdgcn_s_sleep(2);
            }
        }
    }
    __syncthreads();
    __threadfence();
}

// -------------------------- setup kernels ----------------------------------

// Xemb[t][b][h] (bf16) = embed[idx[b][t]][h]
__global__ void __launch_bounds__(256)
embed_gather_kernel(const int* __restrict__ idx, const float* __restrict__ embed,
                    bf16_t* __restrict__ Xemb) {
    size_t id = (size_t)blockIdx.x * blockDim.x + threadIdx.x;  // < T*B*H
    int h = (int)(id & (H_ - 1));
    size_t r = id >> 10;
    int b = (int)(r & (B_ - 1));
    int t = (int)(r >> 6);
    int tok = idx[(size_t)b * T_ + t];
    Xemb[id] = f2bf(embed[(size_t)tok * H_ + h]);
}

// Repack row-major fp32 W (K x N) into the WMMA 32x16 bf16 B-tile layout:
//   out[((nt*numKC + kc)*32 + lane)*16 + e],  lane = (k%16/8)*16 + n%16, e = k%8 + (k%32/16)*8
__global__ void __launch_bounds__(256)
pack_w_kernel(const float* __restrict__ W, bf16_t* __restrict__ out, int K, int N) {
    size_t id = (size_t)blockIdx.x * blockDim.x + threadIdx.x;
    if (id >= (size_t)K * N) return;
    int k = (int)(id / N), n = (int)(id % N);
    int nt = n >> 4, nl = n & 15;
    int kc = k >> 5, kk = k & 31;
    int half = (kk >> 3) & 1, e = (kk & 7) | ((kk >> 4) << 3);
    int lane = half * 16 + nl;
    int numKC = K >> 5;
    out[(((size_t)nt * numKC + kc) * 32 + lane) * 16 + e] = f2bf(W[id]);
}

// zero h/c state and barrier words
__global__ void __launch_bounds__(256)
init_state_kernel(float* c0, float* c1, bf16_t* h0, bf16_t* h1, unsigned* bar) {
    int id = blockIdx.x * blockDim.x + threadIdx.x;   // 65536 threads
    c0[id] = 0.0f; c1[id] = 0.0f;
    h0[id] = f2bf(0.0f); h1[id] = f2bf(0.0f);
    if (id < 8) bar[id] = 0u;
}

// -------------------------- WMMA micro-kernels -----------------------------

// Whole-WG GEMM phase for z = [x | h] @ [WiP ; WhP] (64 x 4096, K=2048).
// The 64-row A slab for each K-chunk group is staged once per WG into LDS
// (double buffered, one __syncthreads per KGROUP chunks); each of the 8 waves
// owns a (nt, mhalf) strip: 2 row-tiles x 16 cols, 2 WMMAs per K-chunk.
__device__ __forceinline__ void wg_gemm_phase(
        const bf16_t* __restrict__ xsrc, const bf16_t* __restrict__ hsrc,
        const bf16_t* __restrict__ WiP,  const bf16_t* __restrict__ WhP,
        float* __restrict__ zout, char* smem,
        int nt, int mhalf, int lane, int tid) {
    v8f acc0 = {};
    v8f acc1 = {};
    const int ml   = lane & 15;
    const int half = lane >> 4;
    const int m0 = mhalf * 32;
    const int m1 = m0 + 16;
    const int srow = tid >> 2;        // staging: row 0..63
    const int sseg = tid & 3;         // staging: 16B segment 0..3

    // 16 groups of KGROUP chunks: groups 0..7 -> x@Wi, 8..15 -> h@Wh.
    // unroll 1: keeps only one group's B fragments live (no scratch spills).
#pragma unroll 1
    for (int g = 0; g < 16; ++g) {
        const bf16_t* As = (g < 8) ? xsrc : hsrc;
        const bf16_t* Bp = (g < 8) ? WiP : WhP;
        const int kkg = (g & 7) * KGROUP;              // chunk idx within half
        char* buf = smem + (g & 1) * (KGROUP * SLAB);
        // ---- stage A slab: KGROUP chunks x 64 rows x 64B (each thread 4x16B)
        // load all segments first (one clause, one wait), then store to LDS
        v8bf stg[KGROUP];
#pragma unroll
        for (int s = 0; s < KGROUP; ++s) {
            stg[s] = *reinterpret_cast<const v8bf*>(
                As + (size_t)srow * H_ + (kkg + s) * 32 + sseg * 8);
        }
#pragma unroll
        for (int s = 0; s < KGROUP; ++s) {
            *reinterpret_cast<v8bf*>(buf + s * SLAB + srow * ROWPAD + sseg * 16) = stg[s];
        }
        __syncthreads();   // double buffer: one barrier per group is sufficient
        // ---- compute KGROUP chunks
#pragma unroll
        for (int s = 0; s < KGROUP; ++s) {
            const int kk = kkg + s;
            const bf16_t* bp = Bp + ((((size_t)nt * 32 + kk) * 32 + lane) << 4);
            if (kk < 31) __builtin_prefetch(bp + (32u << 4), 0, 3);  // near-scope prefetch
            const v16bf bv = *reinterpret_cast<const v16bf*>(bp);
            const char* a0 = buf + s * SLAB + (m0 + ml) * ROWPAD + half * 16;
            const char* a1 = buf + s * SLAB + (m1 + ml) * ROWPAD + half * 16;
            v8bf a0lo = *reinterpret_cast<const v8bf*>(a0);
            v8bf a0hi = *reinterpret_cast<const v8bf*>(a0 + 32);
            v8bf a1lo = *reinterpret_cast<const v8bf*>(a1);
            v8bf a1hi = *reinterpret_cast<const v8bf*>(a1 + 32);
            v16bf av0 = __builtin_shufflevector(a0lo, a0hi, 0,1,2,3,4,5,6,7,8,9,10,11,12,13,14,15);
            v16bf av1 = __builtin_shufflevector(a1lo, a1hi, 0,1,2,3,4,5,6,7,8,9,10,11,12,13,14,15);
            acc0 = __builtin_amdgcn_wmma_f32_16x16x32_bf16(false, av0, false, bv,
                                                           (short)0, acc0, false, false);
            acc1 = __builtin_amdgcn_wmma_f32_16x16x32_bf16(false, av1, false, bv,
                                                           (short)0, acc1, false, false);
        }
        // fence instruction motion at group boundary: bounds live ranges so
        // the scheduler cannot hoist future groups' B loads (spill guard)
        __builtin_amdgcn_sched_barrier(0);
    }
    const int n = nt * 16 + ml;
#pragma unroll
    for (int r = 0; r < 8; ++r) {
        zout[(size_t)(m0 + r + half * 8) * G4_ + n] = acc0[r];
        zout[(size_t)(m1 + r + half * 8) * G4_ + n] = acc1[r];
    }
}

// gates: c/h update, 4 elements per thread (grid = 16384 threads, 65536 elems)
__device__ __forceinline__ void wave_gates(
        const float* __restrict__ z, const float* __restrict__ bias,
        float* __restrict__ c, bf16_t* __restrict__ hbf, int gtid) {
#pragma unroll
    for (int e = 0; e < 4; ++e) {
        int idx = gtid + e * GRID_THREADS;
        int b = idx >> 10;
        int j = idx & (H_ - 1);
        const float* zr = z + (size_t)b * G4_;
        float gi = sigf(zr[j]            + bias[j]);
        float gf = sigf(zr[H_ + j]       + bias[H_ + j]);
        float gg = tanhf(zr[2 * H_ + j]  + bias[2 * H_ + j]);
        float go = sigf(zr[3 * H_ + j]   + bias[3 * H_ + j]);
        float cn = gf * c[idx] + gi * gg;
        c[idx] = cn;
        hbf[idx] = f2bf(go * tanhf(cn));
    }
}

// per-step vocab projection: h1 (64x1024 bf16) @ Wproj -> d_out[b][t][:]
// (divergent per-wave: must contain no __syncthreads)
__device__ __forceinline__ void wave_proj(
        const bf16_t* __restrict__ h1, const bf16_t* __restrict__ WpP,
        float* __restrict__ out, int t, int mt, int nt, int lane) {
    v8f acc = {};
    const int ml = lane & 15, half = lane >> 4;
    const int m0 = mt * 16;
#pragma unroll 4
    for (int kc = 0; kc < 32; ++kc) {
        const v16bf bv = *reinterpret_cast<const v16bf*>(
            WpP + ((((size_t)nt * 32 + kc) * 32 + lane) << 4));
        const bf16_t* ap = h1 + (size_t)(m0 + ml) * H_ + kc * 32 + half * 8;
        v8bf lo = *reinterpret_cast<const v8bf*>(ap);
        v8bf hi = *reinterpret_cast<const v8bf*>(ap + 16);
        v16bf av = __builtin_shufflevector(lo, hi, 0,1,2,3,4,5,6,7,8,9,10,11,12,13,14,15);
        acc = __builtin_amdgcn_wmma_f32_16x16x32_bf16(false, av, false, bv,
                                                      (short)0, acc, false, false);
    }
    const int n = nt * 16 + ml;
#pragma unroll
    for (int r = 0; r < 8; ++r) {
        int b = m0 + r + half * 8;
        out[((size_t)b * T_ + t) * V_ + n] = acc[r];
    }
}

// -------------------------- persistent recurrence --------------------------
__global__ void __launch_bounds__(NTHR, 1)
lstm_persistent_kernel(const bf16_t* __restrict__ Xemb,
                       const bf16_t* __restrict__ WiP0, const bf16_t* __restrict__ WhP0,
                       const bf16_t* __restrict__ WiP1, const bf16_t* __restrict__ WhP1,
                       const bf16_t* __restrict__ WpP,
                       const float*  __restrict__ bias,   // (2, 4096)
                       float* zws, float* c0, float* c1,
                       bf16_t* h0, bf16_t* h1,
                       unsigned* barwords,
                       float* out) {
    __shared__ __align__(16) char smem[SMEM_BYTES];
    const int tid   = threadIdx.x;
    const int lane  = tid & 31;
    const int w     = blockIdx.x * NWAVE_WG + (tid >> 5);  // 0..511
    const int gtid  = blockIdx.x * NTHR + tid;
    const int nt    = w >> 1;       // 0..255 column tile of 4096
    const int mhalf = w & 1;        // rows 0..31 or 32..63
    unsigned* bar = barwords + 0;
    unsigned* gen = barwords + 1;
    unsigned target = 0;

#pragma unroll 1
    for (int t = 0; t < T_; ++t) {
        const bf16_t* xt = Xemb + (size_t)t * (B_ * H_);
        // layer 0: z = x_t @ Wi0 + h0 @ Wh0
        wg_gemm_phase(xt, h0, WiP0, WhP0, zws, smem, nt, mhalf, lane, tid);
        grid_sync(bar, gen, ++target);
        wave_gates(zws, bias, c0, h0, gtid);
        grid_sync(bar, gen, ++target);
        // layer 1: z = h0_t @ Wi1 + h1 @ Wh1
        wg_gemm_phase(h0, h1, WiP1, WhP1, zws, smem, nt, mhalf, lane, tid);
        grid_sync(bar, gen, ++target);
        wave_gates(zws, bias + G4_, c1, h1, gtid);
        grid_sync(bar, gen, ++target);
        // fused projection of this step's h1 into logits (no barrier needed:
        // next step's barriers dominate it before h1 is rewritten)
        if (w < 64) wave_proj(h1, WpP, out, t, w >> 4, w & 15, lane);
    }
}

// -------------------------- host launcher ----------------------------------
extern "C" void kernel_launch(void* const* d_in, const int* in_sizes, int n_in,
                              void* d_out, int out_size, void* d_ws, size_t ws_size,
                              hipStream_t stream) {
    (void)in_sizes; (void)n_in; (void)out_size; (void)ws_size;
    const int*   idx   = (const int*)  d_in[0];   // (64, 512)
    const float* embed = (const float*)d_in[1];   // (256, 1024)
    const float* Wi    = (const float*)d_in[2];   // (2, 1024, 4096)
    const float* Wh    = (const float*)d_in[3];   // (2, 1024, 4096)
    const float* bias  = (const float*)d_in[4];   // (2, 4096)
    const float* Wproj = (const float*)d_in[5];   // (1024, 256)
    float* out = (float*)d_out;                   // (64, 512, 256)

    char* ws = (char*)d_ws;
    size_t off = 0;
    auto alloc = [&](size_t bytes) -> void* {
        off = (off + 255) & ~(size_t)255;
        void* p = ws + off;
        off += bytes;
        return p;
    };
    bf16_t* Xemb = (bf16_t*)alloc((size_t)T_ * B_ * H_ * 2);   // 64 MB
    bf16_t* WiP0 = (bf16_t*)alloc((size_t)H_ * G4_ * 2);       // 8 MB each
    bf16_t* WhP0 = (bf16_t*)alloc((size_t)H_ * G4_ * 2);
    bf16_t* WiP1 = (bf16_t*)alloc((size_t)H_ * G4_ * 2);
    bf16_t* WhP1 = (bf16_t*)alloc((size_t)H_ * G4_ * 2);
    bf16_t* WpP  = (bf16_t*)alloc((size_t)H_ * V_ * 2);
    float*  zws  = (float*) alloc((size_t)B_ * G4_ * 4);       // 1 MB
    float*  c0   = (float*) alloc((size_t)B_ * H_ * 4);
    float*  c1   = (float*) alloc((size_t)B_ * H_ * 4);
    bf16_t* h0   = (bf16_t*)alloc((size_t)B_ * H_ * 2);
    bf16_t* h1   = (bf16_t*)alloc((size_t)B_ * H_ * 2);
    unsigned* barwords = (unsigned*)alloc(256);

    // 1) embedding gather + bf16 cast (time-major)
    embed_gather_kernel<<<(T_ * B_ * H_) / 256, 256, 0, stream>>>(idx, embed, Xemb);
    // 2) weight repack into WMMA B-tile layout
    const int wblk = (H_ * G4_) / 256;
    pack_w_kernel<<<wblk, 256, 0, stream>>>(Wi,                        WiP0, H_, G4_);
    pack_w_kernel<<<wblk, 256, 0, stream>>>(Wi + (size_t)H_ * G4_,     WiP1, H_, G4_);
    pack_w_kernel<<<wblk, 256, 0, stream>>>(Wh,                        WhP0, H_, G4_);
    pack_w_kernel<<<wblk, 256, 0, stream>>>(Wh + (size_t)H_ * G4_,     WhP1, H_, G4_);
    pack_w_kernel<<<(H_ * V_) / 256, 256, 0, stream>>>(Wproj,          WpP,  H_, V_);
    // 3) zero state + barrier words (every call: launcher must be stateless)
    init_state_kernel<<<(B_ * H_) / 256, 256, 0, stream>>>(c0, c1, h0, h1, barwords);
    // 4) persistent recurrent kernel
    lstm_persistent_kernel<<<NWG, NTHR, 0, stream>>>(
        Xemb, WiP0, WhP0, WiP1, WhP1, WpP, bias,
        zws, c0, c1, h0, h1, barwords, out);
}
